// LSTM_28853590295063
// MI455X (gfx1250) — matrix operand order repaired and checked by reading, not compiled, not run on hardware
//
#include <hip/hip_runtime.h>
#include <hip/hip_bf16.h>

// ---------------------------------------------------------------------------
// LSTM (B=32, T=512, IN=1024, H=1024) for gfx1250 using bf16 WMMA.
// Phase 1: x_proj = inputs @ W_ih^T + (b_ih + b_hh)   (big parallel GEMM)
// Phase 2: 512 sequential step kernels:
//          preact = x_proj[:,t,:] + h @ W_hh^T ; gates ; c,h update (fused)
// ---------------------------------------------------------------------------

typedef __bf16 bf16_t;
typedef __attribute__((ext_vector_type(16))) __bf16 v16bf;
typedef __attribute__((ext_vector_type(8)))  float  v8f;

constexpr int B  = 32;
constexpr int T  = 512;
constexpr int IN = 1024;
constexpr int H  = 1024;
constexpr int G4 = 4 * H;          // 4096 gate width
constexpr int BT = B * T;          // 16384 rows of x

// ---- workspace layout (bytes) ---------------------------------------------
constexpr size_t WS_XBF   = 0;                                   // B*T*IN bf16
constexpr size_t WS_WIH   = WS_XBF  + (size_t)BT * IN * 2;       // 4H*IN bf16
constexpr size_t WS_WHH   = WS_WIH  + (size_t)G4 * IN * 2;       // 4H*H  bf16
constexpr size_t WS_HBUF0 = WS_WHH  + (size_t)G4 * H  * 2;       // B*H bf16
constexpr size_t WS_HBUF1 = WS_HBUF0 + (size_t)B * H * 2;        // B*H bf16
constexpr size_t WS_CST   = WS_HBUF1 + (size_t)B * H * 2;        // B*H f32
constexpr size_t WS_XPROJ = WS_CST   + (size_t)B * H * 4;        // B*T*4H f32
// total ~ 304 MB

// ---- fragment loaders (ISA 7.12.2 layouts, wave32) ------------------------
// A: 16x32 bf16, lane(0-15)=row, K 0..7 & 16..23 ; lane(16-31)=row, K 8..15 & 24..31
__device__ __forceinline__ v16bf load_frag_a(const bf16_t* __restrict__ p,
                                             int ld, int row0, int k0) {
  const int lane = threadIdx.x & 31;
  const int r  = row0 + (lane & 15);
  const int kb = k0 + ((lane >> 4) << 3);          // +8 for upper half-wave
  const bf16_t* q = p + (size_t)r * ld + kb;
  union { v16bf v; uint4 u[2]; } f;
  f.u[0] = *(const uint4*)(q);                     // K = kb .. kb+7
  f.u[1] = *(const uint4*)(q + 16);                // K = kb+16 .. kb+23
  return f.v;
}

// B: 32x16 bf16, lane(0-15)=col holds K 0..7 & 16..23 ; lane(16-31)=col holds K 8..15 & 24..31.
// B[k][n] = W[n][k] (row-major W, contiguous in k) -> preact = h @ W^T.
__device__ __forceinline__ v16bf load_frag_b(const bf16_t* __restrict__ w,
                                             int ld, int nrow0, int k0) {
  const int lane = threadIdx.x & 31;
  const int n  = nrow0 + (lane & 15);
  const int kb = k0 + ((lane >> 4) << 3);          // +8 for upper half-wave
  const bf16_t* q = w + (size_t)n * ld + kb;
  union { v16bf v; uint4 u[2]; } f;
  f.u[0] = *(const uint4*)(q);                     // K = kb .. kb+7
  f.u[1] = *(const uint4*)(q + 16);                // K = kb+16 .. kb+23
  return f.v;
}

__device__ __forceinline__ v8f v8f_zero() {
  v8f z;
#pragma unroll
  for (int i = 0; i < 8; ++i) z[i] = 0.0f;
  return z;
}

__device__ __forceinline__ float sigmoidf_(float x) {
  return 1.0f / (1.0f + __expf(-x));
}

// ---- misc elementwise kernels ---------------------------------------------
__global__ void cvt_f32_to_bf16(const float* __restrict__ src,
                                bf16_t* __restrict__ dst, int n) {
  int i = blockIdx.x * blockDim.x + threadIdx.x;
  const int stride = gridDim.x * blockDim.x;
  for (; i < n; i += stride) dst[i] = (bf16_t)src[i];
}

__global__ void init_state(const float* __restrict__ h0,
                           const float* __restrict__ c0,
                           bf16_t* __restrict__ hbuf,
                           float* __restrict__ cstate, int n) {
  int i = blockIdx.x * blockDim.x + threadIdx.x;
  if (i < n) {
    hbuf[i]   = (bf16_t)h0[i];
    cstate[i] = c0[i];
  }
}

// ---- Phase 1: x_proj GEMM --------------------------------------------------
// wave -> one 16-row tile x four 16-col tiles (A reuse). 1024*64 waves total.
__global__ __launch_bounds__(128) void xproj_gemm(
    const bf16_t* __restrict__ xbf, const bf16_t* __restrict__ wih,
    const float* __restrict__ bih, const float* __restrict__ bhh,
    float* __restrict__ xproj) {
  const int wid = threadIdx.x >> 5;
  const int gw  = blockIdx.x * 4 + wid;   // 0 .. 65535
  const int mt  = gw >> 6;                // 0 .. 1023 (16-row tile of B*T)
  const int ng  = gw & 63;                // 0 .. 63   (64-col group)
  const int row0 = mt * 16;
  const int n0   = ng * 64;

  v8f acc[4];
#pragma unroll
  for (int s = 0; s < 4; ++s) acc[s] = v8f_zero();

  for (int kk = 0; kk < IN; kk += 32) {
    v16bf a = load_frag_a(xbf, IN, row0, kk);
#pragma unroll
    for (int s = 0; s < 4; ++s) {
      v16bf b = load_frag_b(wih, IN, n0 + s * 16, kk);
      acc[s] = __builtin_amdgcn_wmma_f32_16x16x32_bf16(
          false, a, false, b, (short)0, acc[s], false, false);
    }
  }

  const int lane = threadIdx.x & 31;
  const int mofs = (lane >> 4) << 3;      // rows 0-7 / 8-15
  const int nl   = lane & 15;
#pragma unroll
  for (int s = 0; s < 4; ++s) {
    const int n = n0 + s * 16 + nl;
    const float bias = bih[n] + bhh[n];
#pragma unroll
    for (int r = 0; r < 8; ++r) {
      const int m = row0 + r + mofs;
      xproj[(size_t)m * G4 + n] = acc[s][r] + bias;
    }
  }
}

// ---- Phase 2: one LSTM time step -------------------------------------------
// 128 waves (grid 32 x 128 threads). Wave -> 16 batch rows x 16 h-cols,
// computing the i,f,o,g tiles for that patch (4 accumulators, shared A).
__global__ __launch_bounds__(128) void lstm_step(
    const bf16_t* __restrict__ hprev, bf16_t* __restrict__ hnext,
    const bf16_t* __restrict__ whh, const float* __restrict__ xproj,
    float* __restrict__ cstate, float* __restrict__ logits, int t) {
  const int wid = threadIdx.x >> 5;
  const int gw  = blockIdx.x * 4 + wid;   // 0 .. 127
  const int mt  = gw >> 6;                // 0..1  (batch rows 0-15 / 16-31)
  const int ht  = gw & 63;                // 0..63 (h column tile)
  const int row0 = mt * 16;
  const int h0c  = ht * 16;

  v8f acc[4];
#pragma unroll
  for (int g = 0; g < 4; ++g) acc[g] = v8f_zero();

  for (int kk = 0; kk < H; kk += 32) {
    v16bf a = load_frag_a(hprev, H, row0, kk);
#pragma unroll
    for (int g = 0; g < 4; ++g) {
      v16bf b = load_frag_b(whh, H, g * H + h0c, kk);
      acc[g] = __builtin_amdgcn_wmma_f32_16x16x32_bf16(
          false, a, false, b, (short)0, acc[g], false, false);
    }
  }

  const int lane = threadIdx.x & 31;
  const int mofs = (lane >> 4) << 3;
  const int n    = h0c + (lane & 15);
#pragma unroll
  for (int r = 0; r < 8; ++r) {
    const int bb = row0 + r + mofs;                      // batch index
    const size_t xrow = ((size_t)bb * T + t) * (size_t)G4;
    const float pi = acc[0][r] + xproj[xrow + 0 * H + n];
    const float pf = acc[1][r] + xproj[xrow + 1 * H + n];
    const float po = acc[2][r] + xproj[xrow + 2 * H + n];
    const float pg = acc[3][r] + xproj[xrow + 3 * H + n];

    const float i_t = sigmoidf_(pi);
    const float f_t = sigmoidf_(pf);
    const float o_t = sigmoidf_(po);
    const float g_t = tanhf(pg);

    const size_t sidx = (size_t)bb * H + n;
    const float c_new = cstate[sidx] * f_t + i_t * g_t;
    const float h_new = o_t * tanhf(c_new);

    cstate[sidx] = c_new;
    logits[((size_t)bb * T + t) * H + n] = h_new;
    hnext[sidx] = (bf16_t)h_new;
  }
}

// ---- tail: h_f, c_f ---------------------------------------------------------
__global__ void write_tail(const float* __restrict__ logits,
                           const float* __restrict__ cstate,
                           float* __restrict__ out_h,
                           float* __restrict__ out_c) {
  int i = blockIdx.x * blockDim.x + threadIdx.x;
  if (i < B * H) {
    const int bb = i / H, n = i % H;
    out_h[i] = logits[((size_t)bb * T + (T - 1)) * H + n];
    out_c[i] = cstate[i];
  }
}

// ---------------------------------------------------------------------------
extern "C" void kernel_launch(void* const* d_in, const int* in_sizes, int n_in,
                              void* d_out, int out_size, void* d_ws, size_t ws_size,
                              hipStream_t stream) {
  const float* inputs = (const float*)d_in[0];   // (B,T,IN)
  const float* h0     = (const float*)d_in[1];   // (1,B,H)
  const float* c0     = (const float*)d_in[2];   // (1,B,H)
  const float* W_ih   = (const float*)d_in[3];   // (4H,IN)
  const float* b_ih   = (const float*)d_in[4];   // (4H,)
  const float* W_hh   = (const float*)d_in[5];   // (4H,H)
  const float* b_hh   = (const float*)d_in[6];   // (4H,)

  char* ws = (char*)d_ws;
  bf16_t* xbf    = (bf16_t*)(ws + WS_XBF);
  bf16_t* wihbf  = (bf16_t*)(ws + WS_WIH);
  bf16_t* whhbf  = (bf16_t*)(ws + WS_WHH);
  bf16_t* hbuf[2] = { (bf16_t*)(ws + WS_HBUF0), (bf16_t*)(ws + WS_HBUF1) };
  float*  cstate = (float*)(ws + WS_CST);
  float*  xproj  = (float*)(ws + WS_XPROJ);

  float* logits = (float*)d_out;                 // (B,T,H)
  float* out_h  = logits + (size_t)B * T * H;    // (1,B,H)
  float* out_c  = out_h + (size_t)B * H;         // (1,B,H)

  // 0) one-time precision conversion into workspace
  cvt_f32_to_bf16<<<4096, 256, 0, stream>>>(inputs, xbf, BT * IN);
  cvt_f32_to_bf16<<<2048, 256, 0, stream>>>(W_ih, wihbf, G4 * IN);
  cvt_f32_to_bf16<<<2048, 256, 0, stream>>>(W_hh, whhbf, G4 * H);
  init_state<<<(B * H + 255) / 256, 256, 0, stream>>>(h0, c0, hbuf[0], cstate, B * H);

  // 1) input projection (bandwidth-bound, full-grid parallel)
  xproj_gemm<<<(BT / 16) * (G4 / 64) / 4, 128, 0, stream>>>(
      xbf, wihbf, b_ih, b_hh, xproj);

  // 2) sequential recurrence; W_hh (8MB bf16) stays L2-resident across steps
  for (int t = 0; t < T; ++t) {
    lstm_step<<<32, 128, 0, stream>>>(hbuf[t & 1], hbuf[(t + 1) & 1],
                                      whhbf, xproj, cstate, logits, t);
  }

  // 3) final states
  write_tail<<<(B * H + 255) / 256, 256, 0, stream>>>(logits, cstate, out_h, out_c);
}
